// MambaBlock_63196148793903
// MI455X (gfx1250) — matrix operand order repaired and checked by reading, not compiled, not run on hardware
//
#include <hip/hip_runtime.h>
#include <math.h>
#include <stdint.h>

// ---------------------------------------------------------------------------
// Mamba block forward for MI455X (gfx1250, wave32).
// GEMMs: bf16x3 split precision (aH*bH + aH*bL + aL*bH, f32 accumulate) on
// V_WMMA_F32_16X16X32_BF16 -- ~fp32 accuracy at the bf16 matrix rate.
// Tiles stream into LDS via GLOBAL_LOAD_ASYNC_TO_LDS_B128 (ASYNCcnt),
// double-buffered. Packing of activations is fused into conv / scan.
// ---------------------------------------------------------------------------

typedef __attribute__((ext_vector_type(8)))  float  v8f;
typedef __attribute__((ext_vector_type(16))) __bf16 v16bf;

#define DMODEL 1024
#define DSTATE 16
#define DCONV  4
#define DINNER 2048
#define BATCH  4
#define SEQ    2048
#define ROWS   (BATCH * SEQ)   // 8192 token rows

#define LROW 20                // LDS row stride in dwords (16 data + 4 pad)

union Frag { v16bf v; uint4 q[2]; };

__device__ __forceinline__ float sigmoidf_(float x) {
    return 1.0f / (1.0f + __expf(-x));
}

// f32 -> (hi,lo) bf16 split by truncation; residual captured exactly.
__device__ __forceinline__ void split_bf16(float a, uint32_t& h, uint32_t& l) {
    uint32_t u  = __float_as_uint(a);
    uint32_t hw = u & 0xFFFF0000u;
    float    r  = a - __uint_as_float(hw);
    h = u >> 16;
    l = __float_as_uint(r) >> 16;
}
__device__ __forceinline__ uint32_t pack2(float a0, float a1, uint32_t& lo) {
    uint32_t h0, l0, h1, l1;
    split_bf16(a0, h0, l0);
    split_bf16(a1, h1, l1);
    lo = l0 | (l1 << 16);
    return h0 | (h1 << 16);
}

// ---- gfx1250 async global->LDS copy (ASYNCcnt) ----------------------------
__device__ __forceinline__ void async_b128(uint32_t lds_byte_addr,
                                           const uint32_t* g) {
    asm volatile("global_load_async_to_lds_b128 %0, %1, off"
                 :: "v"(lds_byte_addr), "v"((uint64_t)(uintptr_t)g)
                 : "memory");
}
#define WAIT_ASYNC(n) asm volatile("s_wait_asynccnt %0" :: "i"(n) : "memory")

// ---------------------------------------------------------------------------
// bf16x3 GEMM: C[M,N] = A[M,K] @ B[K,N] (logically f32).
//   Ah/Al: [M, K/2] dwords, dword = {bf16 a[m][2k] | bf16 a[m][2k+1]<<16}
//   Bh/Bl: [N, K/2] dwords, dword = {bf16 b[2k][n] | bf16 b[2k+1][n]<<16}
// Block tile 128x64, BK=32, 256 threads (8 waves), wave tile 32x32.
// EPI: 0 = plain store, 1 = softplus(v + bias[col]).
// ---------------------------------------------------------------------------
template<int EPI>
__global__ __launch_bounds__(256) void gemm_bf16x3(
    const uint32_t* __restrict__ Ah, const uint32_t* __restrict__ Al,
    const uint32_t* __restrict__ Bh, const uint32_t* __restrict__ Bl,
    const float* __restrict__ bias, float* __restrict__ C,
    int M, int N, int K)
{
    __shared__ uint32_t AsH[2][128 * LROW];
    __shared__ uint32_t AsL[2][128 * LROW];
    __shared__ uint32_t BsH[2][64 * LROW];
    __shared__ uint32_t BsL[2][64 * LROW];

    const int tid  = threadIdx.x;
    const int lane = tid & 31;
    const int wave = tid >> 5;
    const int l16  = lane & 15;
    const int kk   = lane >> 4;
    const int wm   = wave & 3;        // 4 waves tile M (32 rows each)
    const int wn   = wave >> 2;       // 2 waves tile N (32 cols each)
    const int m0   = blockIdx.y * 128;
    const int n0   = blockIdx.x * 64;
    const int K2   = K >> 1;

    v8f acc[2][2] = {};

    auto issue = [&](int kt, int buf) {
        const int kb = kt << 4;       // tile base in dwords
        #pragma unroll
        for (int i = 0; i < 2; ++i) { // A: 512 b128 chunks per matrix
            int c = tid + (i << 8);
            int r = c >> 2, q = (c & 3) << 2;
            size_t   goff = (size_t)(m0 + r) * K2 + kb + q;
            uint32_t loff = (uint32_t)(r * LROW + q) * 4u;
            async_b128((uint32_t)(uintptr_t)AsH[buf] + loff, Ah + goff);
            async_b128((uint32_t)(uintptr_t)AsL[buf] + loff, Al + goff);
        }
        {                             // B: 256 b128 chunks per matrix
            int r = tid >> 2, q = (tid & 3) << 2;
            size_t   goff = (size_t)(n0 + r) * K2 + kb + q;
            uint32_t loff = (uint32_t)(r * LROW + q) * 4u;
            async_b128((uint32_t)(uintptr_t)BsH[buf] + loff, Bh + goff);
            async_b128((uint32_t)(uintptr_t)BsL[buf] + loff, Bl + goff);
        }
    };

    const int T = K >> 5;
    issue(0, 0);
    for (int kt = 0; kt < T; ++kt) {
        const int buf = kt & 1;
        if (kt + 1 < T) { issue(kt + 1, buf ^ 1); WAIT_ASYNC(6); }
        else            { WAIT_ASYNC(0); }
        __syncthreads();              // tile kt visible to all waves

        Frag aH[2], aL[2], bH[2], bL[2];
        #pragma unroll
        for (int mi = 0; mi < 2; ++mi) {
            const uint32_t* pH = &AsH[buf][(wm * 32 + mi * 16 + l16) * LROW];
            const uint32_t* pL = &AsL[buf][(wm * 32 + mi * 16 + l16) * LROW];
            aH[mi].q[0] = *(const uint4*)(pH + kk * 4);
            aH[mi].q[1] = *(const uint4*)(pH + 8 + kk * 4);
            aL[mi].q[0] = *(const uint4*)(pL + kk * 4);
            aL[mi].q[1] = *(const uint4*)(pL + 8 + kk * 4);
        }
        #pragma unroll
        for (int ni = 0; ni < 2; ++ni) {
            const uint32_t* pH = &BsH[buf][(wn * 32 + ni * 16 + l16) * LROW];
            const uint32_t* pL = &BsL[buf][(wn * 32 + ni * 16 + l16) * LROW];
            bH[ni].q[0] = *(const uint4*)(pH + kk * 4);
            bH[ni].q[1] = *(const uint4*)(pH + 8 + kk * 4);
            bL[ni].q[0] = *(const uint4*)(pL + kk * 4);
            bL[ni].q[1] = *(const uint4*)(pL + 8 + kk * 4);
        }
        #pragma unroll
        for (int mi = 0; mi < 2; ++mi)
            #pragma unroll
            for (int ni = 0; ni < 2; ++ni) {
                acc[mi][ni] = __builtin_amdgcn_wmma_f32_16x16x32_bf16(
                    false, aH[mi].v, false, bH[ni].v, (short)0, acc[mi][ni], false, false);
                acc[mi][ni] = __builtin_amdgcn_wmma_f32_16x16x32_bf16(
                    false, aH[mi].v, false, bL[ni].v, (short)0, acc[mi][ni], false, false);
                acc[mi][ni] = __builtin_amdgcn_wmma_f32_16x16x32_bf16(
                    false, aL[mi].v, false, bH[ni].v, (short)0, acc[mi][ni], false, false);
            }
        __syncthreads();
    }

    #pragma unroll
    for (int mi = 0; mi < 2; ++mi)
        #pragma unroll
        for (int ni = 0; ni < 2; ++ni)
            #pragma unroll
            for (int r = 0; r < 8; ++r) {
                const int row = m0 + wm * 32 + mi * 16 + (kk << 3) + r;
                const int col = n0 + wn * 32 + ni * 16 + l16;
                float v = acc[mi][ni][r];
                if (EPI == 1) {
                    v += bias[col];
                    v = fmaxf(v, 0.0f) + log1pf(__expf(-fabsf(v)));
                }
                C[(size_t)row * N + col] = v;
            }
}

// ---------------------------------------------------------------------------
// pack_rows: X[M,K] -> hi/lo[M][K/2] (for hidden_states only).
// ---------------------------------------------------------------------------
__global__ __launch_bounds__(256) void pack_rows_kernel(
    const float* __restrict__ X, uint32_t* __restrict__ hi,
    uint32_t* __restrict__ lo)
{
    const size_t i = (size_t)blockIdx.x * 256 + threadIdx.x;
    uint32_t l;
    uint32_t h = pack2(X[2 * i], X[2 * i + 1], l);
    hi[i] = h;
    lo[i] = l;
}

// ---------------------------------------------------------------------------
// pack_cols: B[K,N] -> hi/lo[N][K/2] (transpose + K-pair pack).
// Each thread: 8 K-rows of one column -> one uint4 store (16B granularity).
// idx over N*(K/8): n = idx%N, k8 = idx/N.
// ---------------------------------------------------------------------------
__global__ __launch_bounds__(256) void pack_cols_kernel(
    const float* __restrict__ B, uint32_t* __restrict__ hi,
    uint32_t* __restrict__ lo, int K, int N)
{
    const size_t idx = (size_t)blockIdx.x * 256 + threadIdx.x;
    const int n  = (int)(idx % N);
    const int k8 = (int)(idx / N);
    float a[8];
    #pragma unroll
    for (int j = 0; j < 8; ++j)
        a[j] = B[(size_t)(k8 * 8 + j) * N + n];
    uint4 hq, lq;
    hq.x = pack2(a[0], a[1], lq.x);
    hq.y = pack2(a[2], a[3], lq.y);
    hq.z = pack2(a[4], a[5], lq.z);
    hq.w = pack2(a[6], a[7], lq.w);
    const size_t o = (size_t)n * (K >> 1) + (size_t)k8 * 4;
    *reinterpret_cast<uint4*>(hi + o) = hq;
    *reinterpret_cast<uint4*>(lo + o) = lq;
}

// ---------------------------------------------------------------------------
// Depthwise causal conv1d (K=4) + SiLU, fused with bf16 hi/lo packing.
// One thread handles a pair of adjacent channels (one packed dword).
// ---------------------------------------------------------------------------
__global__ __launch_bounds__(256) void conv_silu_pack_kernel(
    const float* __restrict__ xz, const float* __restrict__ cw,
    const float* __restrict__ cb, float* __restrict__ xc,
    uint32_t* __restrict__ xcH, uint32_t* __restrict__ xcL)
{
    const size_t idx = (size_t)blockIdx.x * 256 + threadIdx.x; // ROWS*DINNER/2
    const int dp = (int)(idx % (DINNER / 2));   // channel pair
    const size_t bl = idx / (DINNER / 2);       // b*SEQ + l
    const int l = (int)(bl % SEQ);
    const int di = dp * 2;

    float acc0 = cb[di], acc1 = cb[di + 1];
    #pragma unroll
    for (int k = 0; k < DCONV; ++k) {
        const int ll = l + k - (DCONV - 1);
        if (ll >= 0) {
            const size_t row = bl + (size_t)(long long)(k - (DCONV - 1));
            const float2 xv =
                *reinterpret_cast<const float2*>(&xz[row * (2 * DINNER) + di]);
            acc0 += xv.x * cw[di * DCONV + k];
            acc1 += xv.y * cw[(di + 1) * DCONV + k];
        }
    }
    acc0 *= sigmoidf_(acc0);
    acc1 *= sigmoidf_(acc1);

    *reinterpret_cast<float2*>(&xc[bl * DINNER + di]) = make_float2(acc0, acc1);
    uint32_t lw;
    uint32_t hw = pack2(acc0, acc1, lw);
    xcH[bl * (DINNER / 2) + dp] = hw;
    xcL[bl * (DINNER / 2) + dp] = lw;
}

// ---------------------------------------------------------------------------
// BC = xc @ W_x (N=32): one wave per token row, one lane per output column.
// ---------------------------------------------------------------------------
__global__ __launch_bounds__(256) void bc_gemv_kernel(
    const float* __restrict__ xc, const float* __restrict__ Wx,
    float* __restrict__ BC)
{
    const int wave = threadIdx.x >> 5;
    const int lane = threadIdx.x & 31;
    const size_t m = (size_t)blockIdx.x * 8 + wave;
    const float* xr = &xc[m * DINNER];
    float acc = 0.0f;
    #pragma unroll 4
    for (int k = 0; k < DINNER; ++k)
        acc = fmaf(xr[k], Wx[(size_t)k * 32 + lane], acc);
    BC[m * 32 + lane] = acc;
}

// ---------------------------------------------------------------------------
// Selective scan, fused with bf16 hi/lo packing of y.
// lane = (channel, state); lanes 0 and 16 (s==0) hold an adjacent channel
// pair, so lane 0 packs both via one shuffle.  grid (DINNER/16, BATCH).
// ---------------------------------------------------------------------------
__global__ __launch_bounds__(256) void scan_pack_kernel(
    const float* __restrict__ xc, const float* __restrict__ delta,
    const float* __restrict__ bc, const float* __restrict__ xz,
    const float* __restrict__ A_log, const float* __restrict__ Dp,
    uint32_t* __restrict__ yH, uint32_t* __restrict__ yL)
{
    const int tid = threadIdx.x;
    const int s   = tid & 15;
    const int lane = tid & 31;
    const int di  = blockIdx.x * 16 + (tid >> 4);
    const int b   = blockIdx.y;

    const float As = -__expf(A_log[di * DSTATE + s]);
    const float Dv = Dp[di];

    float h = 0.0f;
    for (int l = 0; l < SEQ; ++l) {
        const size_t base = (size_t)b * SEQ + l;
        const float dt = delta[base * DINNER + di];
        const float x  = xc[base * DINNER + di];
        const float Bv = bc[base * 32 + s];
        const float Cv = bc[base * 32 + 16 + s];

        h = h * __expf(dt * As) + (dt * Bv) * x;

        float p = h * Cv;
        p += __shfl_xor(p, 1, 32);
        p += __shfl_xor(p, 2, 32);
        p += __shfl_xor(p, 4, 32);
        p += __shfl_xor(p, 8, 32);

        // y on every lane (all 16 lanes of a channel carry its true value)
        const float z  = xz[base * (2 * DINNER) + DINNER + di];
        const float yv = (p + Dv * x) * z * sigmoidf_(z);
        const float y1 = __shfl(yv, 16, 32);   // odd channel's value
        if (lane == 0) {
            uint32_t lw;
            uint32_t hw = pack2(yv, y1, lw);
            const size_t o = base * (DINNER / 2) + (di >> 1);
            yH[o] = hw;
            yL[o] = lw;
        }
    }
}

// ---------------------------------------------------------------------------
// inputs: 0 hidden_states, 1 W_in, 2 conv_w, 3 conv_b, 4 W_x, 5 W_dt,
//         6 b_dt, 7 A_log, 8 D, 9 W_out
// ---------------------------------------------------------------------------
extern "C" void kernel_launch(void* const* d_in, const int* in_sizes, int n_in,
                              void* d_out, int out_size, void* d_ws, size_t ws_size,
                              hipStream_t stream) {
    const float* hs     = (const float*)d_in[0];
    const float* W_in   = (const float*)d_in[1];
    const float* conv_w = (const float*)d_in[2];
    const float* conv_b = (const float*)d_in[3];
    const float* W_x    = (const float*)d_in[4];
    const float* W_dt   = (const float*)d_in[5];
    const float* b_dt   = (const float*)d_in[6];
    const float* A_log  = (const float*)d_in[7];
    const float* Dp     = (const float*)d_in[8];
    const float* W_out  = (const float*)d_in[9];
    float* out = (float*)d_out;

    char* wsp = (char*)d_ws;
    auto alloc_f32 = [&](size_t n) { float* p = (float*)wsp;       wsp += n * 4; return p; };
    auto alloc_u32 = [&](size_t n) { uint32_t* p = (uint32_t*)wsp; wsp += n * 4; return p; };

    float* xz    = alloc_f32((size_t)ROWS * 2 * DINNER);
    float* xc    = alloc_f32((size_t)ROWS * DINNER);
    float* delta = alloc_f32((size_t)ROWS * DINNER);
    float* bc    = alloc_f32((size_t)ROWS * 2 * DSTATE);

    uint32_t* hsH  = alloc_u32((size_t)ROWS * (DMODEL / 2));
    uint32_t* hsL  = alloc_u32((size_t)ROWS * (DMODEL / 2));
    uint32_t* WinH = alloc_u32((size_t)(2 * DINNER) * (DMODEL / 2));
    uint32_t* WinL = alloc_u32((size_t)(2 * DINNER) * (DMODEL / 2));
    uint32_t* xcH  = alloc_u32((size_t)ROWS * (DINNER / 2));
    uint32_t* xcL  = alloc_u32((size_t)ROWS * (DINNER / 2));
    uint32_t* WdtH = alloc_u32((size_t)DINNER * (DINNER / 2));
    uint32_t* WdtL = alloc_u32((size_t)DINNER * (DINNER / 2));
    uint32_t* yH   = alloc_u32((size_t)ROWS * (DINNER / 2));
    uint32_t* yL   = alloc_u32((size_t)ROWS * (DINNER / 2));
    uint32_t* WoH  = alloc_u32((size_t)DMODEL * (DINNER / 2));
    uint32_t* WoL  = alloc_u32((size_t)DMODEL * (DINNER / 2));

    const dim3 blk(256);

    // 1) pack GEMM1 inputs, xz = hs @ W_in
    pack_rows_kernel<<<(ROWS * (DMODEL / 2)) / 256, blk, 0, stream>>>(hs, hsH, hsL);
    pack_cols_kernel<<<((2 * DINNER) * (DMODEL / 8)) / 256, blk, 0, stream>>>(
        W_in, WinH, WinL, DMODEL, 2 * DINNER);
    gemm_bf16x3<0><<<dim3((2 * DINNER) / 64, ROWS / 128), blk, 0, stream>>>(
        hsH, hsL, WinH, WinL, nullptr, xz, ROWS, 2 * DINNER, DMODEL);

    // 2) depthwise conv + silu (+ fused pack of xc)
    conv_silu_pack_kernel<<<(ROWS * (DINNER / 2)) / 256, blk, 0, stream>>>(
        xz, conv_w, conv_b, xc, xcH, xcL);

    // 3) delta = softplus(xc @ W_dt + b_dt)
    pack_cols_kernel<<<(DINNER * (DINNER / 8)) / 256, blk, 0, stream>>>(
        W_dt, WdtH, WdtL, DINNER, DINNER);
    gemm_bf16x3<1><<<dim3(DINNER / 64, ROWS / 128), blk, 0, stream>>>(
        xcH, xcL, WdtH, WdtL, b_dt, delta, ROWS, DINNER, DINNER);

    // 4) BC = xc @ W_x
    bc_gemv_kernel<<<ROWS / 8, blk, 0, stream>>>(xc, W_x, bc);

    // 5) selective scan (+ fused pack of y)
    scan_pack_kernel<<<dim3(DINNER / 16, BATCH), blk, 0, stream>>>(
        xc, delta, bc, xz, A_log, Dp, yH, yL);

    // 6) out = y @ W_out
    pack_cols_kernel<<<(DMODEL * (DINNER / 8)) / 256, blk, 0, stream>>>(
        W_out, WoH, WoL, DINNER, DMODEL);
    gemm_bf16x3<0><<<dim3(DMODEL / 64, ROWS / 128), blk, 0, stream>>>(
        yH, yL, WoH, WoL, nullptr, out, ROWS, DMODEL, DINNER);
}
